// GroupedQueryAttention_5755256177084
// MI455X (gfx1250) — matrix operand order repaired and checked by reading, not compiled, not run on hardware
//
#include <hip/hip_runtime.h>
#include <hip/hip_bf16.h>

// ---------------------------------------------------------------------------
// GQA forward for MI455X (gfx1250, wave32, WMMA bf16 -> f32 accum)
// ---------------------------------------------------------------------------

typedef __bf16  v16bf __attribute__((ext_vector_type(16)));
typedef float   v8f   __attribute__((ext_vector_type(8)));
typedef unsigned int u32x4 __attribute__((ext_vector_type(4)));
typedef int     i32x4 __attribute__((ext_vector_type(4)));

static_assert(sizeof(v16bf) == 32, "v16bf must be 32 bytes");
static_assert(sizeof(v8f)   == 32, "v8f must be 32 bytes");

union FragBF {            // one WMMA 16-lane-striped operand (8 VGPRs)
    u32x4 q[2];
    v16bf v;
};

__device__ __forceinline__ v8f zero8() {
    v8f z = {0.f, 0.f, 0.f, 0.f, 0.f, 0.f, 0.f, 0.f};
    return z;
}

#define WMMA_BF16(A, B, C) \
    __builtin_amdgcn_wmma_f32_16x16x32_bf16(false, (A), false, (B), (short)0, (C), false, false)

// CDNA5 async global->LDS path (ASYNCcnt), guarded so the build can't regress.
#if __has_builtin(__builtin_amdgcn_global_load_async_to_lds_b128) && \
    __has_builtin(__builtin_amdgcn_s_wait_asynccnt)
#define USE_ASYNC_LDS 1
// Builtin signature (from hipcc diagnostic): (int4 AS1*, int4 AS3*, Imm, Imm)
typedef __attribute__((address_space(1))) i32x4 gl_i32x4;   // global
typedef __attribute__((address_space(3))) i32x4 ld_i32x4;   // LDS
#else
#define USE_ASYNC_LDS 0
#endif

// Problem constants
#define D_MODEL     2048
#define T_SEQ       2048
#define BATCH       2
#define NQ_HEADS    32
#define NKV_HEADS   8
#define HEAD_SZ     64
#define GROUP_SZ    4
#define M_ROWS      (BATCH * T_SEQ)   // 4096

// ---------------------------------------------------------------------------
// fp32 -> bf16 elementwise convert (for X)
// ---------------------------------------------------------------------------
__global__ void gqa_convert_bf16(const float* __restrict__ in,
                                 __bf16* __restrict__ out, int n) {
    int i = blockIdx.x * blockDim.x + threadIdx.x;
    int stride = gridDim.x * blockDim.x;
    for (; i < n; i += stride) out[i] = (__bf16)in[i];
}

// ---------------------------------------------------------------------------
// W [K=2048, N] fp32 row-major  ->  WT [N, 2048] bf16 row-major
// ---------------------------------------------------------------------------
__global__ void gqa_transpose_bf16(const float* __restrict__ in,
                                   __bf16* __restrict__ outT, int N, int total) {
    int i = blockIdx.x * blockDim.x + threadIdx.x;
    if (i < total) {
        int k = i / N;
        int n = i - k * N;
        outT[(size_t)n * D_MODEL + k] = (__bf16)in[i];
    }
}

// ---------------------------------------------------------------------------
// GEMM: Y[M,N] = A[M,2048] * Bt[N,2048]^T   (bf16 in, f32 accum, bf16 out)
// Block: 256 thr = 8 waves. Block tile 128x128, wave tile 32x64, K-step 32.
// Double-buffered LDS, one barrier per K-step. Stage loads use the CDNA5
// async global->LDS path (ASYNCcnt) when available.
// mode 0: Q  row-major [4096, 2048]
// mode 1: K  -> [b, h_kv, t, d]
// mode 2: V  -> [b, h_kv, d, t]   (transposed for P@V B-frags)
// ---------------------------------------------------------------------------
__global__ __launch_bounds__(256) void gqa_gemm_bf16(
        const __bf16* __restrict__ A, const __bf16* __restrict__ Bt,
        __bf16* __restrict__ outp, int mode)
{
    // 80-byte LDS rows: bank stride 20 mod 64 -> conflict-free b128 frag loads
    __shared__ __align__(16) __bf16 lA[2][128][40];
    __shared__ __align__(16) __bf16 lB[2][128][40];

    const int K  = D_MODEL;
    const int t  = threadIdx.x;
    const int lane = t & 31;
    const int lo = lane & 15;       // n / m within a 16-tile
    const int lh = lane >> 4;       // lane half
    const int w  = t >> 5;          // wave 0..7
    const int wm = w & 3;           // wave m position (4 x 32 rows)
    const int wn = w >> 2;          // wave n position (2 x 64 cols)
    const int m0 = blockIdx.y * 128;
    const int n0 = blockIdx.x * 128;

    const int lrow = t >> 1;          // 0..127 (tile row this thread stages)
    const int lseg = (t & 1) * 16;    // 0 or 16 (half-row, 32 bytes)

    const __bf16* aRow = A  + (size_t)(m0 + lrow) * K + lseg;
    const __bf16* bRow = Bt + (size_t)(n0 + lrow) * K + lseg;

    v8f acc[2][4];
#pragma unroll
    for (int mt = 0; mt < 2; ++mt)
#pragma unroll
        for (int nt = 0; nt < 4; ++nt) acc[mt][nt] = zero8();

    auto compute_stage = [&](int s) {
        FragBF fa[2], fb[4];
#pragma unroll
        for (int mt = 0; mt < 2; ++mt) {
            const u32x4* r = (const u32x4*)&lA[s][wm * 32 + mt * 16 + lo][0];
            fa[mt].q[0] = r[lh];        // K = lh*8 + 0..7
            fa[mt].q[1] = r[2 + lh];    // K = 16 + lh*8 + 0..7
        }
#pragma unroll
        for (int nt = 0; nt < 4; ++nt) {
            const u32x4* r = (const u32x4*)&lB[s][wn * 64 + nt * 16 + lo][0];
            fb[nt].q[0] = r[lh];           // K = lh*8 + 0..7
            fb[nt].q[1] = r[2 + lh];       // K = 16 + lh*8 + 0..7
        }
#pragma unroll
        for (int mt = 0; mt < 2; ++mt)
#pragma unroll
            for (int nt = 0; nt < 4; ++nt)
                acc[mt][nt] = WMMA_BF16(fa[mt].v, fb[nt].v, acc[mt][nt]);
    };

    const int nk = K / 32;   // 64 K-steps

#if USE_ASYNC_LDS
    // ---- CDNA5 async copy pipeline: global_load_async_to_lds_b128 ----
    auto issue_stage = [&](int k0, int s) {
        __builtin_amdgcn_global_load_async_to_lds_b128(
            (gl_i32x4*)(aRow + k0),     (ld_i32x4*)&lA[s][lrow][lseg],     0, 0);
        __builtin_amdgcn_global_load_async_to_lds_b128(
            (gl_i32x4*)(aRow + k0 + 8), (ld_i32x4*)&lA[s][lrow][lseg + 8], 0, 0);
        __builtin_amdgcn_global_load_async_to_lds_b128(
            (gl_i32x4*)(bRow + k0),     (ld_i32x4*)&lB[s][lrow][lseg],     0, 0);
        __builtin_amdgcn_global_load_async_to_lds_b128(
            (gl_i32x4*)(bRow + k0 + 8), (ld_i32x4*)&lB[s][lrow][lseg + 8], 0, 0);
    };

    issue_stage(0, 0);
    for (int i = 0; i < nk; ++i) {
        __builtin_amdgcn_s_wait_asynccnt(0);   // stage i resident in LDS
        __syncthreads();                       // all waves done with buf (i+1)&1
        if (i + 1 < nk) issue_stage((i + 1) * 32, (i + 1) & 1);  // overlap
        compute_stage(i & 1);
    }
#else
    // ---- Fallback: register-staged double buffering, 1 barrier per step ----
    u32x4 ra0, ra1, rb0, rb1;
    auto load_gmem = [&](int k0) {
        const u32x4* ga = (const u32x4*)(aRow + k0);
        const u32x4* gb = (const u32x4*)(bRow + k0);
        ra0 = ga[0]; ra1 = ga[1];
        rb0 = gb[0]; rb1 = gb[1];
        __builtin_prefetch(aRow + k0 + 32, 0, 1);   // global_prefetch_b8
        __builtin_prefetch(bRow + k0 + 32, 0, 1);
    };
    auto store_lds = [&](int s) {
        *(u32x4*)&lA[s][lrow][lseg]     = ra0;
        *(u32x4*)&lA[s][lrow][lseg + 8] = ra1;
        *(u32x4*)&lB[s][lrow][lseg]     = rb0;
        *(u32x4*)&lB[s][lrow][lseg + 8] = rb1;
    };

    load_gmem(0);
    store_lds(0);
    __syncthreads();
    for (int i = 0; i < nk; ++i) {
        if (i + 1 < nk) load_gmem((i + 1) * 32);   // overlaps compute below
        compute_stage(i & 1);
        if (i + 1 < nk) {
            store_lds((i + 1) & 1);
            __syncthreads();
        }
    }
#endif

    // Epilogue: D layout -> lane holds n = lo, rows e + 8*lh
#pragma unroll
    for (int mt = 0; mt < 2; ++mt)
#pragma unroll
        for (int nt = 0; nt < 4; ++nt)
#pragma unroll
            for (int e = 0; e < 8; ++e) {
                int m = m0 + wm * 32 + mt * 16 + e + 8 * lh;
                int n = n0 + wn * 64 + nt * 16 + lo;
                __bf16 v = (__bf16)acc[mt][nt][e];
                if (mode == 0) {
                    outp[(size_t)m * D_MODEL + n] = v;
                } else {
                    int bb = m >> 11, tt = m & (T_SEQ - 1);
                    int hh = n >> 6,  dd = n & (HEAD_SZ - 1);
                    if (mode == 1)  // K: [b, h, t, d]
                        outp[(((size_t)(bb * NKV_HEADS + hh) * T_SEQ + tt) << 6) + dd] = v;
                    else            // V: [b, h, d, t]
                        outp[(((size_t)(bb * NKV_HEADS + hh) * HEAD_SZ + dd) << 11) + tt] = v;
                }
            }
}

// ---------------------------------------------------------------------------
// Flash attention: block = 128 thr (4 waves), block handles 64 q rows of one
// (b, h). Each wave owns 16 q rows x full head (64). Online softmax; P is
// re-laid-out D->A via wave-private LDS (DS ops are in-order per wave).
// ---------------------------------------------------------------------------
__global__ __launch_bounds__(128) void gqa_attn(
        const __bf16* __restrict__ Q,    // [4096, 2048] (b,t) x (h,d)
        const __bf16* __restrict__ Kc,   // [b, h_kv, t, d]
        const __bf16* __restrict__ Vt,   // [b, h_kv, d, t]
        float* __restrict__ out)         // [b, t, 2048]
{
    __shared__ __align__(16) __bf16 ldsP[4][16][40];   // per-wave P tile

    const int t    = threadIdx.x;
    const int lane = t & 31;
    const int lo   = lane & 15;
    const int lh   = lane >> 4;
    const int w    = t >> 5;
    const int b    = blockIdx.z;
    const int h    = blockIdx.y;
    const int hk   = h / GROUP_SZ;
    const int qw   = blockIdx.x * 64 + w * 16;    // wave's first q row

    // Q A-frags (16 q rows x 64 d), resident for the whole kernel
    FragBF qf[2];
    const u32x4* qrow =
        (const u32x4*)(Q + ((size_t)(b * T_SEQ + qw + lo)) * D_MODEL + h * HEAD_SZ);
    qf[0].q[0] = qrow[lh];       qf[0].q[1] = qrow[2 + lh];   // d 0..31
    qf[1].q[0] = qrow[4 + lh];   qf[1].q[1] = qrow[6 + lh];   // d 32..63

    float mi[8], li[8];
    v8f o[4];
#pragma unroll
    for (int e = 0; e < 8; ++e) { mi[e] = -1e30f; li[e] = 0.f; }
#pragma unroll
    for (int nt = 0; nt < 4; ++nt) o[nt] = zero8();

    const __bf16* Kbase = Kc + (size_t)(b * NKV_HEADS + hk) * T_SEQ * HEAD_SZ;
    const __bf16* Vbase = Vt + (size_t)(b * NKV_HEADS + hk) * HEAD_SZ * T_SEQ;

    for (int j0 = 0; j0 <= qw + 15; j0 += 32) {    // causal key-tile loop
        // ---- S = Q @ K^T (16 q x 32 keys), f32 accum ----
        v8f s[2];
#pragma unroll
        for (int nt = 0; nt < 2; ++nt) {
            const u32x4* kr =
                (const u32x4*)(Kbase + (size_t)(j0 + nt * 16 + lo) * HEAD_SZ);
            FragBF f0, f1;
            f0.q[0] = kr[lh];     f0.q[1] = kr[2 + lh];       // d 0..31
            f1.q[0] = kr[4 + lh]; f1.q[1] = kr[6 + lh];       // d 32..63
            v8f z = zero8();
            z = WMMA_BF16(qf[0].v, f0.v, z);
            z = WMMA_BF16(qf[1].v, f1.v, z);
            s[nt] = z * 0.125f;       // 1/sqrt(64)
        }

        // ---- causal mask (only on diagonal tiles) ----
        if (j0 + 31 > qw) {
#pragma unroll
            for (int nt = 0; nt < 2; ++nt)
#pragma unroll
                for (int e = 0; e < 8; ++e) {
                    int key = j0 + nt * 16 + lo;
                    int row = qw + e + 8 * lh;
                    if (key > row) s[nt][e] = -1e30f;
                }
        }

        // ---- online softmax: per-row max/sum across the 16-lane half ----
#pragma unroll
        for (int e = 0; e < 8; ++e) {
            float mx = fmaxf(s[0][e], s[1][e]);
            mx = fmaxf(mx, __shfl_xor(mx, 1, 32));
            mx = fmaxf(mx, __shfl_xor(mx, 2, 32));
            mx = fmaxf(mx, __shfl_xor(mx, 4, 32));
            mx = fmaxf(mx, __shfl_xor(mx, 8, 32));
            float mn   = fmaxf(mi[e], mx);
            float corr = __expf(mi[e] - mn);
            float p0 = __expf(s[0][e] - mn);
            float p1 = __expf(s[1][e] - mn);
            s[0][e] = p0; s[1][e] = p1;
            float rs = p0 + p1;
            rs += __shfl_xor(rs, 1, 32);
            rs += __shfl_xor(rs, 2, 32);
            rs += __shfl_xor(rs, 4, 32);
            rs += __shfl_xor(rs, 8, 32);
            li[e] = li[e] * corr + rs;
            mi[e] = mn;
#pragma unroll
            for (int nt = 0; nt < 4; ++nt) o[nt][e] *= corr;
        }

        // ---- P (D layout) -> LDS -> A-frag. Wave-private; DS is in-order. ----
        asm volatile("" ::: "memory");
#pragma unroll
        for (int nt = 0; nt < 2; ++nt)
#pragma unroll
            for (int e = 0; e < 8; ++e)
                ldsP[w][e + 8 * lh][nt * 16 + lo] = (__bf16)s[nt][e];
        asm volatile("" ::: "memory");

        FragBF pf;
        {
            const u32x4* pr = (const u32x4*)&ldsP[w][lo][0];
            pf.q[0] = pr[lh];
            pf.q[1] = pr[2 + lh];
        }
        asm volatile("" ::: "memory");

        // ---- O += P @ V (V^T rows are key-contiguous -> clean B-frags) ----
#pragma unroll
        for (int nt = 0; nt < 4; ++nt) {
            const u32x4* vr =
                (const u32x4*)(Vbase + (size_t)(nt * 16 + lo) * T_SEQ + j0);
            FragBF vf;
            vf.q[0] = vr[lh];
            vf.q[1] = vr[2 + lh];
            o[nt] = WMMA_BF16(pf.v, vf.v, o[nt]);
        }
    }

    // ---- write O / l, fp32, [b, t, h*64 + d] ----
#pragma unroll
    for (int nt = 0; nt < 4; ++nt)
#pragma unroll
        for (int e = 0; e < 8; ++e) {
            int row = qw + e + 8 * lh;
            int col = h * HEAD_SZ + nt * 16 + lo;
            out[(size_t)(b * T_SEQ + row) * D_MODEL + col] = o[nt][e] / li[e];
        }
}

// ---------------------------------------------------------------------------
// Host launcher
// ---------------------------------------------------------------------------
extern "C" void kernel_launch(void* const* d_in, const int* in_sizes, int n_in,
                              void* d_out, int out_size, void* d_ws, size_t ws_size,
                              hipStream_t stream) {
    (void)in_sizes; (void)n_in; (void)out_size; (void)ws_size;
    const float* x  = (const float*)d_in[0];
    const float* Wq = (const float*)d_in[1];
    const float* Wk = (const float*)d_in[2];
    const float* Wv = (const float*)d_in[3];
    float* out = (float*)d_out;

    char* ws = (char*)d_ws;
    size_t off = 0;
    auto carve = [&](size_t bytes) -> void* {
        void* p = ws + off;
        off += (bytes + 255) & ~(size_t)255;
        return p;
    };
    __bf16* Xb  = (__bf16*)carve((size_t)M_ROWS * D_MODEL * 2);            // 16 MB
    __bf16* WqT = (__bf16*)carve((size_t)D_MODEL * D_MODEL * 2);           // 8 MB
    __bf16* WkT = (__bf16*)carve((size_t)(NKV_HEADS * HEAD_SZ) * D_MODEL * 2);
    __bf16* WvT = (__bf16*)carve((size_t)(NKV_HEADS * HEAD_SZ) * D_MODEL * 2);
    __bf16* Qb  = (__bf16*)carve((size_t)M_ROWS * D_MODEL * 2);            // 16 MB
    __bf16* Kb  = (__bf16*)carve((size_t)M_ROWS * NKV_HEADS * HEAD_SZ);    // bytes
    __bf16* Vt  = (__bf16*)carve((size_t)M_ROWS * NKV_HEADS * HEAD_SZ);

    // 1) precision conversion + weight transposes (bf16, [N,K] layout)
    gqa_convert_bf16<<<4096, 256, 0, stream>>>(x, Xb, M_ROWS * D_MODEL);
    gqa_transpose_bf16<<<(D_MODEL * D_MODEL + 255) / 256, 256, 0, stream>>>(
        Wq, WqT, NQ_HEADS * HEAD_SZ, D_MODEL * NQ_HEADS * HEAD_SZ);
    gqa_transpose_bf16<<<(D_MODEL * 512 + 255) / 256, 256, 0, stream>>>(
        Wk, WkT, NKV_HEADS * HEAD_SZ, D_MODEL * NKV_HEADS * HEAD_SZ);
    gqa_transpose_bf16<<<(D_MODEL * 512 + 255) / 256, 256, 0, stream>>>(
        Wv, WvT, NKV_HEADS * HEAD_SZ, D_MODEL * NKV_HEADS * HEAD_SZ);

    // 2) QKV projections (WMMA bf16)
    gqa_gemm_bf16<<<dim3(D_MODEL / 128, M_ROWS / 128), 256, 0, stream>>>(
        Xb, WqT, Qb, 0);
    gqa_gemm_bf16<<<dim3(512 / 128, M_ROWS / 128), 256, 0, stream>>>(
        Xb, WkT, Kb, 1);
    gqa_gemm_bf16<<<dim3(512 / 128, M_ROWS / 128), 256, 0, stream>>>(
        Xb, WvT, Vt, 2);

    // 3) causal flash attention (WMMA bf16, online softmax)
    gqa_attn<<<dim3(T_SEQ / 64, 32, 2), 128, 0, stream>>>(
        Qb, Kb, Vt, out);
}
